// Transformer_60266981097537
// MI455X (gfx1250) — matrix operand order repaired and checked by reading, not compiled, not run on hardware
//
#include <hip/hip_runtime.h>
#include <hip/hip_bf16.h>
#include <math.h>

typedef __attribute__((ext_vector_type(16))) _Float16 v16h;
typedef __attribute__((ext_vector_type(8)))  float    v8f;

#define MT 128
#define NT 128
#define KTILE 32

#if __has_builtin(__builtin_amdgcn_tensor_load_to_lds) && __has_builtin(__builtin_amdgcn_s_wait_tensorcnt)
#define HAVE_TDM 1
#else
#define HAVE_TDM 0
#endif

__device__ __forceinline__ float gelu_exact(float x) {
    return 0.5f * x * (1.0f + erff(x * 0.70710678118654752f));
}

__device__ __forceinline__ float urand(unsigned x) {
    x ^= x >> 16; x *= 0x7feb352dU; x ^= x >> 15; x *= 0x846ca68bU; x ^= x >> 16;
    return ((float)(x >> 8) + 0.5f) * (1.0f / 16777216.0f);
}

#if HAVE_TDM
typedef __attribute__((ext_vector_type(4))) unsigned int u32x4_t;
typedef __attribute__((ext_vector_type(8))) int          i32x8_t;
typedef __attribute__((ext_vector_type(4))) int          i32x4_t;

// Issue one TDM load of a 2D f16 tile [tileRows x tileK] (row stride ldElems)
// into LDS at ldsDst. remK/remRows are the remaining tensor extents from the
// tile origin: the TDM zero-fills reads beyond tensor_dim, which handles edge
// tiles (K not multiple of 32, M/N not multiple of tile) in hardware.
__device__ __forceinline__ void tdm_load_tile_f16(const _Float16* gsrc, long ldElems,
                                                  unsigned remK, unsigned remRows,
                                                  unsigned tileK, unsigned tileRows,
                                                  void* ldsDst)
{
    unsigned long long ga = (unsigned long long)gsrc;
    unsigned lds = (unsigned)(unsigned long long)ldsDst;   // low 32 bits = LDS byte offset
    unsigned long long s0 = (unsigned long long)ldElems;   // dim0 stride in data_size units

    u32x4_t g0;
    g0.x = 1u;                                             // count=1 (valid), user mode
    g0.y = lds;                                            // lds_addr
    g0.z = (unsigned)(ga & 0xffffffffull);                 // global_addr[31:0]
    g0.w = (unsigned)((ga >> 32) & 0x01ffffffull) | (2u << 30); // addr[56:32] | type=2

    i32x8_t g1;
    g1[0] = (int)(1u << 16);                               // wg_mask=0, data_size=1 (2 bytes)
    g1[1] = (int)((remK & 0xffffu) << 16);                 // tensor_dim0[15:0]
    g1[2] = (int)((remK >> 16) | ((remRows & 0xffffu) << 16)); // td0[31:16] | td1[15:0]
    g1[3] = (int)((remRows >> 16) | (tileK << 16));        // td1[31:16] | tile_dim0
    g1[4] = (int)tileRows;                                 // tile_dim1 (tile_dim2=0)
    g1[5] = (int)(s0 & 0xffffffffull);                     // tensor_dim0_stride[31:0]
    g1[6] = (int)((s0 >> 32) & 0xffffull);                 // stride[47:32], dim1_stride=0
    g1[7] = 0;

    i32x4_t gz4 = {0, 0, 0, 0};                            // D# groups 2/3 (unused for 2D)
    i32x8_t gz8 = {0, 0, 0, 0, 0, 0, 0, 0};
    __builtin_amdgcn_tensor_load_to_lds(g0, g1, gz4, gz4, gz8, 0);
}
#endif

// ---------------------------------------------------------------------------
// Batched NT WMMA GEMM: C = act(A(MxK) * Bt(NxK)^T + bias) + addend
// batch index z decomposed as (z/sub, z%sub) with outer/inner strides.
// act: 0 = none, 1 = gelu
// Tiles staged global->LDS via the Tensor Data Mover (waves 0/1 issue the
// A/B tile DMAs, TENSORcnt-synced), compute via v_wmma_f32_16x16x32_f16.
// ---------------------------------------------------------------------------
__global__ __launch_bounds__(256)
void gemm_wmma_nt(const _Float16* __restrict__ A, long lda, long Ao, long Ai,
                  const _Float16* __restrict__ Bt, long ldb, long Bo, long Bi,
                  float* __restrict__ C, long ldc, long Co, long Ci,
                  const float* __restrict__ bias,
                  const float* __restrict__ addend, long ldadd, long Addo, long Addi,
                  int M, int N, int K, int sub, int act)
{
    __shared__ __align__(16) _Float16 As[2][MT * KTILE];
    __shared__ __align__(16) _Float16 Bs[2][NT * KTILE];

    const int z  = blockIdx.z;
    const int zo = z / sub, zi = z % sub;
    A  += zo * Ao + zi * Ai;
    Bt += zo * Bo + zi * Bi;
    C  += zo * Co + zi * Ci;
    if (addend) addend += zo * Addo + zi * Addi;

    const int tileM = blockIdx.y * MT;
    const int tileN = blockIdx.x * NT;
    const int tid   = threadIdx.x;
    const int lane  = tid & 31;
    const int wid   = tid >> 5;
    const int wr    = wid >> 1;   // 0..3 -> M offset wr*32
    const int wc    = wid & 1;    // 0..1 -> N offset wc*64
    const int lan   = lane & 15;
    const int hsel  = lane >> 4;

#if HAVE_TDM
    auto issueLoads = [&](int kbase, int buf) {
        if (wid == 0) {
            tdm_load_tile_f16(A + (long)tileM * lda + kbase, lda,
                              (unsigned)(K - kbase), (unsigned)(M - tileM),
                              KTILE, MT, &As[buf][0]);
        } else if (wid == 1) {
            tdm_load_tile_f16(Bt + (long)tileN * ldb + kbase, ldb,
                              (unsigned)(K - kbase), (unsigned)(N - tileN),
                              KTILE, NT, &Bs[buf][0]);
        }
    };
#else
    auto loadA = [&](int kbase, int buf) {
        int row = tid >> 1, seg = tid & 1;
        int kk = kbase + seg * 16;
        _Float16* dst = &As[buf][row * KTILE + seg * 16];
        int gr = tileM + row;
        if (gr < M && (kk + 16) <= K) {
            const uint4* p = reinterpret_cast<const uint4*>(A + (long)gr * lda + kk);
            *reinterpret_cast<uint4*>(dst)     = p[0];
            *reinterpret_cast<uint4*>(dst + 8) = p[1];
        } else {
            for (int e = 0; e < 16; ++e)
                dst[e] = (gr < M && (kk + e) < K) ? A[(long)gr * lda + kk + e] : (_Float16)0.f;
        }
    };
    auto loadB = [&](int kbase, int buf) {
        int row = tid >> 1, seg = tid & 1;
        int kk = kbase + seg * 16;
        _Float16* dst = &Bs[buf][row * KTILE + seg * 16];
        int gc = tileN + row;
        if (gc < N && (kk + 16) <= K) {
            const uint4* p = reinterpret_cast<const uint4*>(Bt + (long)gc * ldb + kk);
            *reinterpret_cast<uint4*>(dst)     = p[0];
            *reinterpret_cast<uint4*>(dst + 8) = p[1];
        } else {
            for (int e = 0; e < 16; ++e)
                dst[e] = (gc < N && (kk + e) < K) ? Bt[(long)gc * ldb + kk + e] : (_Float16)0.f;
        }
    };
#endif

    v8f acc[2][4];
#pragma unroll
    for (int mi = 0; mi < 2; ++mi)
#pragma unroll
        for (int ni = 0; ni < 4; ++ni) acc[mi][ni] = (v8f){0.f,0.f,0.f,0.f,0.f,0.f,0.f,0.f};

    const int kTiles = (K + KTILE - 1) / KTILE;

#if HAVE_TDM
    issueLoads(0, 0);
    if (wid < 2) __builtin_amdgcn_s_wait_tensorcnt(0);
#else
    loadA(0, 0); loadB(0, 0);
#endif
    __syncthreads();

    for (int kt = 0; kt < kTiles; ++kt) {
        const int nb = kt & 1;
#if HAVE_TDM
        if (kt + 1 < kTiles) issueLoads((kt + 1) * KTILE, nb ^ 1);
#else
        if (kt + 1 < kTiles) { loadA((kt + 1) * KTILE, nb ^ 1); loadB((kt + 1) * KTILE, nb ^ 1); }
        if (kt + 2 < kTiles) {
            int row = tid >> 1, seg = tid & 1;
            int pr = tileM + row; if (pr >= M) pr = M - 1;
            int pc = tileN + row; if (pc >= N) pc = N - 1;
            __builtin_prefetch(A  + (long)pr * lda + (kt + 2) * KTILE + seg * 16, 0, 1);
            __builtin_prefetch(Bt + (long)pc * ldb + (kt + 2) * KTILE + seg * 16, 0, 1);
        }
#endif

        v16h af[2], bf[4];
#pragma unroll
        for (int mi = 0; mi < 2; ++mi) {
            const _Float16* ap = &As[nb][(wr * 32 + mi * 16 + lan) * KTILE];
#pragma unroll
            for (int j = 0; j < 8; ++j) af[mi][j]     = ap[hsel * 8 + j];
#pragma unroll
            for (int j = 0; j < 8; ++j) af[mi][8 + j] = ap[16 + hsel * 8 + j];
        }
#pragma unroll
        for (int ni = 0; ni < 4; ++ni) {
            const _Float16* bp = &Bs[nb][(wc * 64 + ni * 16 + lan) * KTILE];
#pragma unroll
            for (int j = 0; j < 8; ++j) bf[ni][j]     = bp[hsel * 8 + j];
#pragma unroll
            for (int j = 0; j < 8; ++j) bf[ni][8 + j] = bp[16 + hsel * 8 + j];
        }
#pragma unroll
        for (int mi = 0; mi < 2; ++mi)
#pragma unroll
            for (int ni = 0; ni < 4; ++ni)
                acc[mi][ni] = __builtin_amdgcn_wmma_f32_16x16x32_f16(
                    false, af[mi], false, bf[ni], (short)0, acc[mi][ni], false, false);

#if HAVE_TDM
        if (wid < 2 && kt + 1 < kTiles) __builtin_amdgcn_s_wait_tensorcnt(0);
#endif
        __syncthreads();
    }

    // epilogue: C/D layout -> VGPR r holds M = r + 8*hsel, lane n = lan
#pragma unroll
    for (int mi = 0; mi < 2; ++mi)
#pragma unroll
        for (int ni = 0; ni < 4; ++ni)
#pragma unroll
            for (int r = 0; r < 8; ++r) {
                int m = tileM + wr * 32 + mi * 16 + hsel * 8 + r;
                int n = tileN + wc * 64 + ni * 16 + lan;
                if (m < M && n < N) {
                    float v = acc[mi][ni][r];
                    if (bias) v += bias[n];
                    if (act == 1) v = gelu_exact(v);
                    if (addend) v += addend[(long)m * ldadd + n];
                    C[(long)m * ldc + n] = v;
                }
            }
}

// ---------------------------------------------------------------------------
// Generic strided small matmul (scalar): for tiny N (<=8) ops.
// act: 0 none, 1 gelu, 2 sigmoid
// ---------------------------------------------------------------------------
__global__ void small_mm(const float* __restrict__ A, long am, long ak, long ab,
                         const float* __restrict__ B, long bk, long bn, long bb,
                         const float* __restrict__ A2, long a2m, long a2k, long a2b,
                         const float* __restrict__ B2, long b2k, long b2n, long b2b, int K2,
                         const float* __restrict__ bias, long biasn,
                         const float* __restrict__ divp, long dm, long dn, long db,
                         float* __restrict__ C, long cm, long cn, long cb,
                         int M, int N, int K, int act)
{
    const int z = blockIdx.z;
    long i = blockIdx.x * (long)blockDim.x + threadIdx.x;
    if (i >= (long)M * N) return;
    int m = (int)(i / N), n = (int)(i % N);
    const float* Ap = A + (long)z * ab;
    const float* Bp = B + (long)z * bb;
    float acc = 0.f;
    for (int k = 0; k < K; ++k) acc += Ap[(long)m * am + (long)k * ak] * Bp[(long)k * bk + (long)n * bn];
    if (A2) {
        const float* A2p = A2 + (long)z * a2b;
        const float* B2p = B2 + (long)z * b2b;
        for (int k = 0; k < K2; ++k) acc += A2p[(long)m * a2m + (long)k * a2k] * B2p[(long)k * b2k + (long)n * b2n];
    }
    if (bias) acc += bias[(long)n * biasn];
    if (divp) acc /= divp[(long)z * db + (long)m * dm + (long)n * dn];
    if (act == 1) acc = gelu_exact(acc);
    else if (act == 2) acc = 1.f / (1.f + __expf(-acc));
    C[(long)z * cb + (long)m * cm + (long)n * cn] = acc;
}

// ---------------------------------------------------------------------------
// f32 -> f16 transpose: dst[c*R + r] = src[r*srow + c]; batch offset (z/sub, z%sub)
// ---------------------------------------------------------------------------
__global__ void transpose_f16(const float* __restrict__ src, long srow, long sOut, long sIn, int sub,
                              _Float16* __restrict__ dst, long dbatch, int R, int C)
{
    const int z = blockIdx.z;
    src += (long)(z / sub) * sOut + (long)(z % sub) * sIn;
    dst += (long)z * dbatch;
    for (long i = blockIdx.x * (long)blockDim.x + threadIdx.x; i < (long)R * C;
         i += (long)gridDim.x * blockDim.x) {
        int r = (int)(i / C), c = (int)(i % C);
        dst[(long)c * R + r] = (_Float16)src[(long)r * srow + c];
    }
}

__global__ void f32_to_f16(const float* __restrict__ s, _Float16* __restrict__ d, long n)
{
    for (long i = blockIdx.x * (long)blockDim.x + threadIdx.x; i < n;
         i += (long)gridDim.x * blockDim.x)
        d[i] = (_Float16)s[i];
}

// build block-diagonal Bt[n][k] = (n/48==k/48) ? w[(k%48)*48 + n%48] : 0   (w: 48x48)
__global__ void blockdiag_f16(const float* __restrict__ w, _Float16* __restrict__ dst)
{
    long i = blockIdx.x * (long)blockDim.x + threadIdx.x;
    if (i >= (long)384 * 384) return;
    int n = (int)(i / 384), k = (int)(i % 384);
    float v = (n / 48 == k / 48) ? w[(k % 48) * 48 + (n % 48)] : 0.f;
    dst[i] = (_Float16)v;
}

__global__ void adj_prepare(const int* __restrict__ adj, _Float16* __restrict__ a16,
                            _Float16* __restrict__ aT16, float* __restrict__ maskf, int t)
{
    for (long i = blockIdx.x * (long)blockDim.x + threadIdx.x; i < (long)t * t;
         i += (long)gridDim.x * blockDim.x) {
        int r = (int)(i / t), c = (int)(i % t);
        float v = (float)adj[i];
        a16[i] = (_Float16)v;
        aT16[(long)c * t + r] = (_Float16)v;
        maskf[i] = v;
    }
}

__global__ void layernorm_f16(const float* __restrict__ x, const float* __restrict__ g,
                              const float* __restrict__ bta, _Float16* __restrict__ out, int cols)
{
    const long row = blockIdx.x;
    const float* xr = x + row * cols;
    __shared__ float red[128];
    const int tid = threadIdx.x;
    float s = 0.f;
    for (int c = tid; c < cols; c += 128) s += xr[c];
    red[tid] = s; __syncthreads();
    for (int o = 64; o > 0; o >>= 1) { if (tid < o) red[tid] += red[tid + o]; __syncthreads(); }
    const float mean = red[0] / cols; __syncthreads();
    float v = 0.f;
    for (int c = tid; c < cols; c += 128) { float d = xr[c] - mean; v += d * d; }
    red[tid] = v; __syncthreads();
    for (int o = 64; o > 0; o >>= 1) { if (tid < o) red[tid] += red[tid + o]; __syncthreads(); }
    const float inv = rsqrtf(red[0] / cols + 1e-5f);
    for (int c = tid; c < cols; c += 128)
        out[row * cols + c] = (_Float16)((xr[c] - mean) * inv * g[c] + bta[c]);
}

// mixed[b,l,i,j] = mask( sum_h randmat[l,h] * rmd[b,h,i]*leaky(scale*score[b,h,i,j])*rmd[b,h,j] )
__global__ void score_mix(const float* __restrict__ score, const float* __restrict__ rmd,
                          const float* __restrict__ rmat, const float* __restrict__ maskp,
                          long maskStride, float* __restrict__ out, int t, float scale)
{
    const long total = (long)8 * 8 * t * t;
    for (long i = blockIdx.x * (long)blockDim.x + threadIdx.x; i < total;
         i += (long)gridDim.x * blockDim.x) {
        int j = (int)(i % t);
        long r = i / t;
        int ii = (int)(r % t); r /= t;
        int l  = (int)(r % 8);
        int b  = (int)(r / 8);
        float acc = 0.f;
#pragma unroll
        for (int h = 0; h < 8; ++h) {
            float sc = score[(((long)b * 8 + h) * t + ii) * t + j] * scale;
            sc = sc > 0.f ? sc : 0.01f * sc;
            acc += rmat[l * 8 + h] * rmd[((long)b * 8 + h) * t + ii] * sc *
                   rmd[((long)b * 8 + h) * t + j];
        }
        float mv = maskp[(long)b * maskStride + (long)ii * t + j];
        out[i] = (mv > 0.f) ? acc : -1e12f;
    }
}

__global__ void softmax_rows_f16(const float* __restrict__ in, _Float16* __restrict__ out, int t)
{
    const long row = blockIdx.x;
    const float* r = in + row * t;
    __shared__ float red[256];
    const int tid = threadIdx.x;
    float v = (tid < t) ? r[tid] : -3.0e38f;
    red[tid] = v; __syncthreads();
    for (int o = 128; o > 0; o >>= 1) { if (tid < o) red[tid] = fmaxf(red[tid], red[tid + o]); __syncthreads(); }
    const float m = red[0]; __syncthreads();
    const float e = (tid < t) ? __expf(v - m) : 0.f;
    red[tid] = e; __syncthreads();
    for (int o = 128; o > 0; o >>= 1) { if (tid < o) red[tid] += red[tid + o]; __syncthreads(); }
    if (tid < t) out[row * t + tid] = (_Float16)(e / red[0]);
}

// gumbel softmax over s for tmp[b,s,l]; outT16[b,l,s]
__global__ void gumbel_pool(const float* __restrict__ tmp, _Float16* __restrict__ outT,
                            int t, int t2, unsigned seed)
{
    const int bl = blockIdx.x;
    const int b = bl / t2, l = bl % t2;
    const int s = threadIdx.x;
    __shared__ float red[256];
    float v = -3.0e38f;
    if (s < t) {
        unsigned key = seed ^ ((unsigned)(bl * 977 + s) * 2654435761u + 12345u);
        float u = urand(key);
        v = tmp[((long)b * t + s) * t2 + l] - logf(-logf(u));
    }
    red[s] = v; __syncthreads();
    for (int o = 128; o > 0; o >>= 1) { if (s < o) red[s] = fmaxf(red[s], red[s + o]); __syncthreads(); }
    const float m = red[0]; __syncthreads();
    const float e = (s < t) ? __expf(v - m) : 0.f;
    red[s] = e; __syncthreads();
    for (int o = 128; o > 0; o >>= 1) { if (s < o) red[s] += red[s + o]; __syncthreads(); }
    if (s < t) outT[((long)b * t2 + l) * t + s] = (_Float16)(e / red[0]);
}

// final gumbel-softmax over 64 logits + weighted sum -> out[b, 384]
__global__ void final_pool(const float* __restrict__ lg, const float* __restrict__ x,
                           float* __restrict__ out, unsigned seed)
{
    const int b = blockIdx.x;
    __shared__ float w[64];
    __shared__ float norm;
    const int tid = threadIdx.x;
    if (tid < 64) {
        unsigned key = seed ^ ((unsigned)(b * 64 + tid) * 2654435761u);
        float u = urand(key);
        w[tid] = lg[b * 64 + tid] - logf(-logf(u));
    }
    __syncthreads();
    if (tid == 0) {
        float m = -3.0e38f;
        for (int l = 0; l < 64; ++l) m = fmaxf(m, w[l]);
        float s = 0.f;
        for (int l = 0; l < 64; ++l) { float e = __expf(w[l] - m); w[l] = e; s += e; }
        norm = 1.f / s;
    }
    __syncthreads();
    const float inv = norm;
    for (int n = tid; n < 384; n += blockDim.x) {
        float acc = 0.f;
        for (int l = 0; l < 64; ++l) acc += w[l] * inv * x[((long)b * 64 + l) * 384 + n];
        out[b * 384 + n] = acc;
    }
}

// ---------------------------------------------------------------------------
// patchify + conv1 (3x3 s2 p1) + BN + gelu + depthwise conv2 (3x3 s2 p1)
// + 2x2 maxpool -> Y16[patch, c*4 + py*2 + px]  (2048 x 1536, f16)
// ---------------------------------------------------------------------------
__global__ __launch_bounds__(256)
void patch_embed(const float* __restrict__ img, const float* __restrict__ w1,
                 const float* __restrict__ b1, const float* __restrict__ bng,
                 const float* __restrict__ bnb, const float* __restrict__ w2,
                 const float* __restrict__ b2, _Float16* __restrict__ Y16)
{
    const int pb = blockIdx.x;                 // 0..2047
    const int b = pb >> 8, p = pb & 255, gy = p >> 4, gx = p & 15;
    __shared__ float sIn[3 * 14 * 14];
    __shared__ float sC1[64 * 49];
    __shared__ float sC2[64 * 16];
    const int tid = threadIdx.x;
    for (int i = tid; i < 3 * 14 * 14; i += 256) {
        int c = i / 196, rem = i % 196, py = rem / 14, px = rem % 14;
        sIn[i] = img[(((long)b * 3 + c) * 224 + (gy * 14 + py)) * 224 + gx * 14 + px];
    }
    __syncthreads();
    const float BN_INV = 0.99999500003749969f;
    for (int cb = 0; cb < 384; cb += 64) {
        for (int i = tid; i < 64 * 49; i += 256) {
            int oc = cb + i / 49;
            int oy = (i % 49) / 7, ox = (i % 49) % 7;
            float acc = b1[oc];
            for (int c = 0; c < 3; ++c)
                for (int ky = 0; ky < 3; ++ky) {
                    int iy = oy * 2 - 1 + ky;
                    if ((unsigned)iy < 14u)
                        for (int kx = 0; kx < 3; ++kx) {
                            int ix = ox * 2 - 1 + kx;
                            if ((unsigned)ix < 14u)
                                acc += sIn[c * 196 + iy * 14 + ix] * w1[((oc * 3 + c) * 3 + ky) * 3 + kx];
                        }
                }
            acc = acc * BN_INV * bng[oc] + bnb[oc];
            sC1[i] = gelu_exact(acc);
        }
        __syncthreads();
        for (int i = tid; i < 64 * 16; i += 256) {
            int lc = i / 16, oc = cb + lc;
            int oy = (i % 16) / 4, ox = (i % 16) % 4;
            float acc = b2[oc];
            for (int ky = 0; ky < 3; ++ky) {
                int iy = oy * 2 - 1 + ky;
                if ((unsigned)iy < 7u)
                    for (int kx = 0; kx < 3; ++kx) {
                        int ix = ox * 2 - 1 + kx;
                        if ((unsigned)ix < 7u)
                            acc += sC1[lc * 49 + iy * 7 + ix] * w2[oc * 9 + ky * 3 + kx];
                    }
            }
            sC2[i] = acc;
        }
        __syncthreads();
        for (int i = tid; i < 64 * 4; i += 256) {
            int lc = i / 4, oc = cb + lc;
            int py = (i % 4) / 2, px = (i % 4) % 2;
            float m = fmaxf(fmaxf(sC2[lc * 16 + (2 * py) * 4 + 2 * px],
                                  sC2[lc * 16 + (2 * py) * 4 + 2 * px + 1]),
                            fmaxf(sC2[lc * 16 + (2 * py + 1) * 4 + 2 * px],
                                  sC2[lc * 16 + (2 * py + 1) * 4 + 2 * px + 1]));
            Y16[(long)pb * 1536 + oc * 4 + py * 2 + px] = (_Float16)m;
        }
        __syncthreads();
    }
}

// ---------------------------------------------------------------------------
// host orchestration
// ---------------------------------------------------------------------------
extern "C" void kernel_launch(void* const* d_in, const int* in_sizes, int n_in,
                              void* d_out, int out_size, void* d_ws, size_t ws_size,
                              hipStream_t stream)
{
    (void)in_sizes; (void)n_in; (void)out_size; (void)ws_size;
    const float* img  = (const float*)d_in[0];
    const int*   adj  = (const int*)  d_in[1];
    const float* ln_g = (const float*)d_in[2];
    const float* ln_b = (const float*)d_in[3];
    const float* gwq  = (const float*)d_in[4];
    const float* gwk  = (const float*)d_in[5];
    const float* gwv  = (const float*)d_in[6];
    const float* wq   = (const float*)d_in[7];
    const float* wk   = (const float*)d_in[8];
    const float* wv   = (const float*)d_in[9];
    const float* wkf  = (const float*)d_in[10];
    const float* wkf2 = (const float*)d_in[11];
    const float* wkf3 = (const float*)d_in[12];
    const float* sD   = (const float*)d_in[13];
    const float* sD2  = (const float*)d_in[14];
    const float* rmat = (const float*)d_in[15];
    const float* c1w  = (const float*)d_in[16];
    const float* c1b  = (const float*)d_in[17];
    const float* bng  = (const float*)d_in[18];
    const float* bnb  = (const float*)d_in[19];
    const float* c2w  = (const float*)d_in[20];
    const float* c2b  = (const float*)d_in[21];
    const float* embw = (const float*)d_in[22];
    const float* embb = (const float*)d_in[23];
    const float* upw  = (const float*)d_in[24];
    const float* upb  = (const float*)d_in[25];
    const float* uow  = (const float*)d_in[26];
    const float* uob  = (const float*)d_in[27];

    char* base = (char*)d_ws;
    size_t off = 0;
    auto alloc = [&](size_t bytes) -> char* {
        size_t a = (off + 255) & ~(size_t)255;
        char* p = base + a;
        off = a + bytes;
        return p;
    };

    _Float16* Y16     = (_Float16*)alloc((size_t)2048 * 1536 * 2);
    _Float16* embT16  = (_Float16*)alloc((size_t)384 * 1536 * 2);
    float*    xA      = (float*)   alloc((size_t)2048 * 384 * 4);
    float*    xB      = (float*)   alloc((size_t)2048 * 384 * 4);
    _Float16* x16     = (_Float16*)alloc((size_t)2048 * 384 * 2);
    _Float16* xn16    = (_Float16*)alloc((size_t)2048 * 384 * 2);
    _Float16* Wbdq    = (_Float16*)alloc((size_t)384 * 384 * 2);
    _Float16* Wbdk    = (_Float16*)alloc((size_t)384 * 384 * 2);
    _Float16* Wbdv    = (_Float16*)alloc((size_t)384 * 384 * 2);
    _Float16* wqT16   = (_Float16*)alloc((size_t)384 * 384 * 2);
    _Float16* wkT16   = (_Float16*)alloc((size_t)384 * 384 * 2);
    _Float16* wvT16   = (_Float16*)alloc((size_t)384 * 384 * 2);
    float*    xwq     = (float*)   alloc((size_t)2048 * 384 * 4);
    float*    xwk     = (float*)   alloc((size_t)2048 * 384 * 4);
    float*    xwv     = (float*)   alloc((size_t)2048 * 384 * 4);
    _Float16* xwqT16  = (_Float16*)alloc((size_t)8 * 384 * 256 * 2);
    _Float16* xwkT16  = (_Float16*)alloc((size_t)8 * 384 * 256 * 2);
    _Float16* xwvT16  = (_Float16*)alloc((size_t)64 * 48 * 256 * 2);
    float*    q4f     = (float*)   alloc((size_t)2048 * 384 * 4);
    float*    k4f     = (float*)   alloc((size_t)2048 * 384 * 4);
    _Float16* q416    = (_Float16*)alloc((size_t)2048 * 384 * 2);
    _Float16* k416    = (_Float16*)alloc((size_t)2048 * 384 * 2);
    float*    qf      = (float*)   alloc((size_t)2048 * 384 * 4);
    float*    kf      = (float*)   alloc((size_t)2048 * 384 * 4);
    _Float16* q16     = (_Float16*)alloc((size_t)2048 * 384 * 2);
    _Float16* k16     = (_Float16*)alloc((size_t)2048 * 384 * 2);
    float*    rm0F    = (float*)   alloc((size_t)2048 * 8 * 4);
    float*    rmF     = (float*)   alloc((size_t)8 * 64 * 4);
    float*    rmdF    = (float*)   alloc((size_t)8 * 8 * 256 * 4);
    float*    scoref  = (float*)   alloc((size_t)8 * 8 * 256 * 256 * 4);
    float*    mixedf  = (float*)   alloc((size_t)8 * 8 * 256 * 256 * 4);
    _Float16* attn16  = (_Float16*)alloc((size_t)8 * 8 * 256 * 256 * 2);
    float*    v4f     = (float*)   alloc((size_t)2048 * 384 * 4);
    _Float16* v416    = (_Float16*)alloc((size_t)2048 * 384 * 2);
    _Float16* adj16   = (_Float16*)alloc((size_t)256 * 256 * 2);
    _Float16* adjT16  = (_Float16*)alloc((size_t)256 * 256 * 2);
    float*    maskf   = (float*)   alloc((size_t)256 * 256 * 4);
    _Float16* upT16   = (_Float16*)alloc((size_t)64 * 384 * 2);
    float*    tmpf    = (float*)   alloc((size_t)2048 * 64 * 4);
    _Float16* tmpT16  = (_Float16*)alloc((size_t)8 * 64 * 256 * 2);
    _Float16* xT16    = (_Float16*)alloc((size_t)8 * 384 * 256 * 2);
    float*    G1f     = (float*)   alloc((size_t)8 * 64 * 256 * 4);
    _Float16* G116    = (_Float16*)alloc((size_t)8 * 64 * 256 * 2);
    float*    repadjf = (float*)   alloc((size_t)8 * 64 * 64 * 4);
    _Float16* adj16p  = (_Float16*)alloc((size_t)8 * 64 * 64 * 2);
    float*    lgf     = (float*)   alloc((size_t)512 * 4);

    auto blocks1 = [](long n) { return dim3((unsigned)((n + 255) / 256), 1, 1); };
    auto blocksZ = [](long n, int z) { return dim3((unsigned)((n + 255) / 256), 1, (unsigned)z); };

    auto GEMM = [&](const _Float16* A, long lda, long Ao, long Ai,
                    const _Float16* B, long ldb, long Bo, long Bi,
                    float* C, long ldc, long Co, long Ci,
                    const float* bias, const float* add, long ldadd, long Addo, long Addi,
                    int M, int N, int K, int batch, int sub, int act) {
        dim3 g((unsigned)((N + NT - 1) / NT), (unsigned)((M + MT - 1) / MT), (unsigned)batch);
        gemm_wmma_nt<<<g, 256, 0, stream>>>(A, lda, Ao, Ai, B, ldb, Bo, Bi, C, ldc, Co, Ci,
                                            bias, add, ldadd, Addo, Addi, M, N, K, sub, act);
    };

    const float SCALE = 0.05103103630798288f;  // 384^-0.5

    // ---- patch embedding ----
    patch_embed<<<2048, 256, 0, stream>>>(img, c1w, c1b, bng, bnb, c2w, c2b, Y16);
    transpose_f16<<<blocks1((long)1536 * 384), 256, 0, stream>>>(embw, 384, 0, 0, 1, embT16, 0, 1536, 384);
    GEMM(Y16, 1536, 0, 0, embT16, 1536, 0, 0, xA, 384, 0, 0, embb,
         nullptr, 0, 0, 0, 2048, 384, 1536, 1, 1, 0);
    adj_prepare<<<blocks1((long)256 * 256), 256, 0, stream>>>(adj, adj16, adjT16, maskf, 256);

    float* x = xA;
    float* xAlt = xB;
    const _Float16* adjPtr = adj16;
    long adjOuter = 0;
    const float* maskPtr = maskf;
    long maskStride = 0;

    const int Tt[4] = {256, 256, 64, 64};
    for (int i = 0; i < 4; ++i) {
        const int t = Tt[i];
        const long bt = (long)8 * t;

        if (i == 2) {
            // ---- token pooling (operates on x with t_prev = 256) ----
            f32_to_f16<<<blocks1((long)2048 * 384), 256, 0, stream>>>(x, x16, (long)2048 * 384);
            transpose_f16<<<blocks1((long)384 * 64), 256, 0, stream>>>(upw, 64, 0, 0, 1, upT16, 0, 384, 64);
            GEMM(x16, 384, 0, 0, upT16, 384, 0, 0, tmpf, 64, 0, 0, upb,
                 nullptr, 0, 0, 0, 2048, 64, 384, 1, 1, 0);
            gumbel_pool<<<8 * 64, 256, 0, stream>>>(tmpf, tmpT16, 256, 64, 0xC0FFEEu + i);
            transpose_f16<<<blocksZ((long)256 * 384, 8), 256, 0, stream>>>(
                x, 384, (long)256 * 384, 0, 1, xT16, (long)384 * 256, 256, 384);
            GEMM(tmpT16, 256, (long)64 * 256, 0, xT16, 256, (long)384 * 256, 0,
                 xAlt, 384, (long)64 * 384, 0, nullptr, nullptr, 0, 0, 0, 64, 384, 256, 8, 1, 0);
            { float* tp = x; x = xAlt; xAlt = tp; }
            GEMM(tmpT16, 256, (long)64 * 256, 0, adjT16, 256, 0, 0,
                 G1f, 256, (long)64 * 256, 0, nullptr, nullptr, 0, 0, 0, 64, 256, 256, 8, 1, 0);
            f32_to_f16<<<blocks1((long)8 * 64 * 256), 256, 0, stream>>>(G1f, G116, (long)8 * 64 * 256);
            GEMM(G116, 256, (long)64 * 256, 0, tmpT16, 256, (long)64 * 256, 0,
                 repadjf, 64, (long)64 * 64, 0, nullptr, nullptr, 0, 0, 0, 64, 64, 256, 8, 1, 0);
            f32_to_f16<<<blocks1((long)8 * 64 * 64), 256, 0, stream>>>(repadjf, adj16p, (long)8 * 64 * 64);
            adjPtr = adj16p; adjOuter = (long)64 * 64;
            maskPtr = repadjf; maskStride = (long)64 * 64;
        }

        // ---- layernorm -> f16 ----
        layernorm_f16<<<(unsigned)bt, 128, 0, stream>>>(x, ln_g + i * 384, ln_b + i * 384, xn16, 384);

        // ---- per-layer weight prep ----
        blockdiag_f16<<<blocks1((long)384 * 384), 256, 0, stream>>>(gwq + i * 2304, Wbdq);
        blockdiag_f16<<<blocks1((long)384 * 384), 256, 0, stream>>>(gwk + i * 2304, Wbdk);
        blockdiag_f16<<<blocks1((long)384 * 384), 256, 0, stream>>>(gwv + i * 2304, Wbdv);
        transpose_f16<<<blocks1((long)384 * 384), 256, 0, stream>>>(wq + (long)i * 147456, 384, 0, 0, 1, wqT16, 0, 384, 384);
        transpose_f16<<<blocks1((long)384 * 384), 256, 0, stream>>>(wk + (long)i * 147456, 384, 0, 0, 1, wkT16, 0, 384, 384);
        transpose_f16<<<blocks1((long)384 * 384), 256, 0, stream>>>(wv + (long)i * 147456, 384, 0, 0, 1, wvT16, 0, 384, 384);

        // ---- head projections (block-diagonal) ----
        GEMM(xn16, 384, 0, 0, Wbdq, 384, 0, 0, xwq, 384, 0, 0, nullptr, nullptr, 0, 0, 0, (int)bt, 384, 384, 1, 1, 0);
        GEMM(xn16, 384, 0, 0, Wbdk, 384, 0, 0, xwk, 384, 0, 0, nullptr, nullptr, 0, 0, 0, (int)bt, 384, 384, 1, 1, 0);
        GEMM(xn16, 384, 0, 0, Wbdv, 384, 0, 0, xwv, 384, 0, 0, nullptr, nullptr, 0, 0, 0, (int)bt, 384, 384, 1, 1, 0);

        // ---- GCN mix: q4 = adj @ XWq ; k4 = adj @ XWk (batched over b) ----
        transpose_f16<<<blocksZ((long)t * 384, 8), 256, 0, stream>>>(
            xwq, 384, (long)t * 384, 0, 1, xwqT16, (long)384 * t, t, 384);
        transpose_f16<<<blocksZ((long)t * 384, 8), 256, 0, stream>>>(
            xwk, 384, (long)t * 384, 0, 1, xwkT16, (long)384 * t, t, 384);
        GEMM(adjPtr, t, adjOuter, 0, xwqT16, t, (long)384 * t, 0,
             q4f, 384, (long)t * 384, 0, nullptr, nullptr, 0, 0, 0, t, 384, t, 8, 1, 0);
        GEMM(adjPtr, t, adjOuter, 0, xwkT16, t, (long)384 * t, 0,
             k4f, 384, (long)t * 384, 0, nullptr, nullptr, 0, 0, 0, t, 384, t, 8, 1, 0);
        f32_to_f16<<<blocks1(bt * 384), 256, 0, stream>>>(q4f, q416, bt * 384);
        f32_to_f16<<<blocks1(bt * 384), 256, 0, stream>>>(k4f, k416, bt * 384);

        // ---- q/k projections ----
        GEMM(q416, 384, 0, 0, wqT16, 384, 0, 0, qf, 384, 0, 0, nullptr, nullptr, 0, 0, 0, (int)bt, 384, 384, 1, 1, 0);
        GEMM(k416, 384, 0, 0, wkT16, 384, 0, 0, kf, 384, 0, 0, nullptr, nullptr, 0, 0, 0, (int)bt, 384, 384, 1, 1, 0);
        f32_to_f16<<<blocks1(bt * 384), 256, 0, stream>>>(qf, q16, bt * 384);
        f32_to_f16<<<blocks1(bt * 384), 256, 0, stream>>>(kf, k16, bt * 384);

        // ---- rm0 = gelu([q|k] @ w_kf) ; rm = rm0^T rm0 ; rmd = sigmoid(rm@w_kf{2,3}/sD) ----
        small_mm<<<blocks1(bt * 8), 256, 0, stream>>>(
            qf, 384, 1, 0, wkf + (long)i * 6144, 8, 1, 0,
            kf, 384, 1, 0, wkf + (long)i * 6144 + 384 * 8, 8, 1, 0, 384,
            nullptr, 0, nullptr, 0, 0, 0, rm0F, 8, 1, 0, (int)bt, 8, 384, 1);
        small_mm<<<blocksZ(64, 8), 256, 0, stream>>>(
            rm0F, 1, 8, (long)t * 8, rm0F, 8, 1, (long)t * 8,
            nullptr, 0, 0, 0, nullptr, 0, 0, 0, 0,
            nullptr, 0, nullptr, 0, 0, 0, rmF, 8, 1, 64, 8, 8, t, 0);
        const float* kfB = (i < 2) ? (wkf2 + (long)i * 2048) : (wkf3 + (long)i * 512);
        const float* sDB = (i < 2) ? (sD + (long)i * 2048) : (sD2 + (long)i * 512);
        small_mm<<<blocksZ((long)8 * t, 8), 256, 0, stream>>>(
            rmF, 8, 1, 64, kfB, t, 1, 0,
            nullptr, 0, 0, 0, nullptr, 0, 0, 0, 0,
            nullptr, 0, sDB, t, 1, 0, rmdF, t, 1, (long)8 * t, 8, t, 8, 2);

        // ---- scores: q_h k_h^T (batched over b,h), mix, mask, softmax ----
        GEMM(q16, 384, (long)t * 384, 48, k16, 384, (long)t * 384, 48,
             scoref, t, (long)8 * t * t, (long)t * t, nullptr, nullptr, 0, 0, 0,
             t, t, 48, 64, 8, 0);
        score_mix<<<2048, 256, 0, stream>>>(scoref, rmdF, rmat + i * 64, maskPtr, maskStride,
                                            mixedf, t, SCALE);
        softmax_rows_f16<<<(unsigned)(64 * t), 256, 0, stream>>>(mixedf, attn16, t);

        // ---- v4 = attn @ XWv (batched over b,h), output proj + gelu + residual ----
        transpose_f16<<<blocksZ((long)t * 48, 64), 256, 0, stream>>>(
            xwv, 384, (long)t * 384, 48, 8, xwvT16, (long)48 * t, t, 48);
        GEMM(attn16, t, (long)8 * t * t, (long)t * t, xwvT16, t, (long)8 * 48 * t, (long)48 * t,
             v4f, 384, (long)t * 384, 48, nullptr, nullptr, 0, 0, 0, t, 48, t, 64, 8, 0);
        f32_to_f16<<<blocks1(bt * 384), 256, 0, stream>>>(v4f, v416, bt * 384);
        GEMM(v416, 384, 0, 0, wvT16, 384, 0, 0, xAlt, 384, 0, 0, nullptr,
             x, 384, 0, 0, (int)bt, 384, 384, 1, 1, 1);
        { float* tp = x; x = xAlt; xAlt = tp; }
    }

    // ---- final pooling ----
    small_mm<<<blocks1(512), 256, 0, stream>>>(
        x, 384, 1, 0, uow, 1, 1, 0,
        nullptr, 0, 0, 0, nullptr, 0, 0, 0, 0,
        uob, 0, nullptr, 0, 0, 0, lgf, 1, 1, 0, 512, 1, 384, 0);
    final_pool<<<8, 128, 0, stream>>>(lgf, x, (float*)d_out, 0xDEADBEEFu);
}
